// CrystallineBottleneck_67697274520388
// MI455X (gfx1250) — matrix operand order repaired and checked by reading, not compiled, not run on hardware
//
#include <hip/hip_runtime.h>
#include <hip/hip_bf16.h>

// ---------------------------------------------------------------------------
// CrystallineBottleneck fused kernel for gfx1250 (MI455X, wave32, WMMA)
//
// Outputs concatenated in d_out (float):
//   [0 .. 33554432)            output      (8*4096*1024)
//   [33554432 .. 50331648)     soft_codes  (8*4096*512)
//   [50331648 .. 67108864)     hard_codes  (8*4096*512)
//   [67108864]                 entropy
//   [67108865]                 tau
// ---------------------------------------------------------------------------

#define DIM   1024
#define CBN   512
#define ROWS  32768        // 8 * 4096
#define KSEL  8
#define LDA   1032         // padded LDS row stride for f16 A (16B-aligned, bank-friendly)
#define LDL   516          // padded LDS row stride for f32 logits

typedef __attribute__((ext_vector_type(16))) _Float16 v16h;
typedef __attribute__((ext_vector_type(8)))  _Float16 v8h;
typedef __attribute__((ext_vector_type(8)))  float    v8f;
typedef __attribute__((ext_vector_type(4)))  float    v4f;

// ---------------------------------------------------------------------------
// Kernel 1: normalize codebook rows -> f16 (WMMA B operand feed), reset
// entropy accumulator, and publish tau. 512 blocks x 256 threads.
// ---------------------------------------------------------------------------
__global__ __launch_bounds__(256) void cb_prep_kernel(
    const float* __restrict__ cb, _Float16* __restrict__ cbn,
    const float* __restrict__ temp, float* __restrict__ entOut,
    float* __restrict__ tauOut) {
  __shared__ float red[8];
  const int row = blockIdx.x;
  const int tid = threadIdx.x;
  const float* src = cb + (size_t)row * DIM;

  float ss = 0.f;
  #pragma unroll
  for (int i = 0; i < 4; ++i) {
    float v = src[tid + i * 256];
    ss += v * v;
  }
  // wave32 reduce
  #pragma unroll
  for (int m = 1; m < 32; m <<= 1) ss += __shfl_xor(ss, m, 32);
  if ((tid & 31) == 0) red[tid >> 5] = ss;
  __syncthreads();
  float tot = red[0] + red[1] + red[2] + red[3] +
              red[4] + red[5] + red[6] + red[7];
  float inv = 1.0f / fmaxf(sqrtf(tot), 1e-12f);

  _Float16* dst = cbn + (size_t)row * DIM;
  #pragma unroll
  for (int i = 0; i < 4; ++i) {
    int d = tid + i * 256;
    dst[d] = (_Float16)(src[d] * inv);
  }

  if (row == 0 && tid == 0) {
    *entOut = 0.0f;
    *tauOut = fmaxf(temp[0], 0.1f);
  }
}

// ---------------------------------------------------------------------------
// Kernel 2: fused normalize + WMMA logits + gumbel-softmax + top-8 + output.
// 1024 blocks (32 rows each) x 256 threads (8 waves).
// ---------------------------------------------------------------------------
__global__ __launch_bounds__(256) void bottleneck_kernel(
    const float* __restrict__ x, const float* __restrict__ u,
    const float* __restrict__ cb, const _Float16* __restrict__ cbn,
    const float* __restrict__ oscale, const float* __restrict__ temp,
    float* __restrict__ out, float* __restrict__ soft,
    float* __restrict__ hard, float* __restrict__ entOut) {
  __shared__ _Float16 sA[32 * LDA];   // normalized x rows, f16
  __shared__ float    sL[32 * LDL];   // logits -> p -> exp -> soft (in place)

  const int tid  = threadIdx.x;
  const int r    = tid >> 3;          // local row 0..31
  const int sub  = tid & 7;           // 8 threads per row
  const size_t gr = (size_t)blockIdx.x * 32 + r;

  // ---- Phase 1: L2-normalize 32 x-rows into LDS as f16 -------------------
  {
    const float* xr = x + gr * DIM;
    float ss = 0.f;
    #pragma unroll 8
    for (int i = 0; i < 128; ++i) {
      float v = xr[sub + i * 8];
      ss += v * v;
    }
    #pragma unroll
    for (int m = 1; m < 8; m <<= 1) ss += __shfl_xor(ss, m, 8);
    float inv = 1.0f / fmaxf(sqrtf(ss), 1e-12f);
    _Float16* ar = &sA[r * LDA];
    #pragma unroll 8
    for (int i = 0; i < 128; ++i) {
      int d = sub + i * 8;
      ar[d] = (_Float16)(xr[d] * inv);
    }
  }
  __syncthreads();

  // ---- Phase 2: logits GEMM via v_wmma_f32_16x16x32_f16 ------------------
  // wave w: row tile (w>>2)*16, columns (w&3)*128 .. +128 (8 16x16 tiles)
  {
    const int wave    = tid >> 5;
    const int lane    = tid & 31;
    const int rtile   = wave >> 2;
    const int colbase = (wave & 3) * 128;
    const int m       = lane & 15;
    const int khalf   = lane >> 4;

    v8f acc[8] = {};
    const _Float16* aBase = &sA[(rtile * 16 + m) * LDA + khalf * 16];
    const _Float16* bBase = cbn + (size_t)(colbase + m) * DIM + khalf * 16;

    for (int k0 = 0; k0 < DIM; k0 += 32) {
      // A fragment: lanes 0-15 hold K k0+0..15; lanes 16-31 hold
      // K k0+16..31 (contiguous, matching B).
      v8h lo = *(const v8h*)(aBase + k0);
      v8h hi = *(const v8h*)(aBase + k0 + 8);
      v16h a;
      #pragma unroll
      for (int i = 0; i < 8; ++i) { a[i] = lo[i]; a[i + 8] = hi[i]; }

      const _Float16* bp = bBase + k0;
      __builtin_prefetch((const void*)(bp + 32), 0, 0);  // next K chunk
      #pragma unroll
      for (int t = 0; t < 8; ++t) {
        // B fragment: lane = column (colbase+16t+m), 16 contiguous K values.
        v16h b = *(const v16h*)(bp + (size_t)t * 16 * DIM);
        acc[t] = __builtin_amdgcn_wmma_f32_16x16x32_f16(
            false, a, false, b, (short)0, acc[t], false, false);
      }
    }

    // Spill logits to LDS (C layout: lanes 0-15 N=lane M=j; 16-31 M=j+8).
    #pragma unroll
    for (int t = 0; t < 8; ++t) {
      #pragma unroll
      for (int j = 0; j < 8; ++j) {
        int row_l = rtile * 16 + ((lane < 16) ? j : j + 8);
        int col   = colbase + 16 * t + m;
        sL[row_l * LDL + col] = acc[t][j];
      }
    }
  }
  __syncthreads();

  // ---- Phase 3: gumbel softmax / entropy / top-8 / outputs ---------------
  const float invtau = 1.0f / fmaxf(temp[0], 0.1f);
  const float* ur = u + gr * CBN + sub * 64;
  float* sLr = &sL[r * LDL + sub * 64];

  // perturbed logits p = (logit + gumbel)/tau, track row max
  float mx = -1e30f;
  #pragma unroll 4
  for (int i = 0; i < 64; ++i) {
    float uv = ur[i];
    float g  = -__logf(-__logf(uv + 1e-8f) + 1e-8f);
    float p  = (sLr[i] + g) * invtau;
    sLr[i] = p;
    mx = fmaxf(mx, p);
  }
  #pragma unroll
  for (int m = 1; m < 8; m <<= 1) mx = fmaxf(mx, __shfl_xor(mx, m, 8));

  float sum = 0.f;
  #pragma unroll 4
  for (int i = 0; i < 64; ++i) {
    float e = __expf(sLr[i] - mx);
    sLr[i] = e;
    sum += e;
  }
  #pragma unroll
  for (int m = 1; m < 8; m <<= 1) sum += __shfl_xor(sum, m, 8);
  const float invs = 1.0f / sum;

  float* srow = soft + gr * CBN;
  float ent = 0.f;
  #pragma unroll 4
  for (int i = 0; i < 64; ++i) {
    float s = sLr[i] * invs;
    sLr[i] = s;
    srow[sub * 64 + i] = s;
    ent += s * __logf(s + 1e-8f);
  }
  #pragma unroll
  for (int m = 1; m < 8; m <<= 1) ent += __shfl_xor(ent, m, 8);
  if (sub == 0) atomicAdd(entOut, -ent * (1.0f / (float)ROWS));

  // top-8 (iterative argmax over the row, ties -> lowest index)
  float tv[KSEL];
  int   ti[KSEL];
  float ksum = 0.f;
  #pragma unroll
  for (int k = 0; k < KSEL; ++k) {
    float lv = -1.0f;
    int   li = 0;
    for (int i = 0; i < 64; ++i) {
      float v = sLr[i];
      if (v > lv) { lv = v; li = sub * 64 + i; }
    }
    #pragma unroll
    for (int m = 1; m < 8; m <<= 1) {
      float ov = __shfl_xor(lv, m, 8);
      int   oi = __shfl_xor(li, m, 8);
      if (ov > lv || (ov == lv && oi < li)) { lv = ov; li = oi; }
    }
    tv[k] = lv; ti[k] = li; ksum += lv;
    if ((li >> 6) == sub) sLr[li & 63] = -1.0f;  // exclude from next rounds
  }
  const float invk = 1.0f / ksum;

  // dense hard codes
  float* hrow = hard + gr * CBN;
  for (int i = 0; i < 64; ++i) {
    int c = sub * 64 + i;
    float h = 0.f;
    #pragma unroll
    for (int k = 0; k < KSEL; ++k)
      if (ti[k] == c) h = tv[k] * invk;
    hrow[c] = h;
  }

  // sparse reconstruction: out = scale * sum_k w_k * codebook[idx_k]
  const float scale = oscale[0];
  float w[KSEL];
  const float* cbr[KSEL];
  #pragma unroll
  for (int k = 0; k < KSEL; ++k) {
    w[k]   = tv[k] * invk * scale;
    cbr[k] = cb + (size_t)ti[k] * DIM;
  }
  float* orow = out + gr * DIM;
  for (int i = 0; i < 32; ++i) {
    int d = (i * 8 + sub) * 4;
    v4f o = {0.f, 0.f, 0.f, 0.f};
    #pragma unroll
    for (int k = 0; k < KSEL; ++k) {
      v4f cv = *(const v4f*)(cbr[k] + d);
      o += w[k] * cv;
    }
    *(v4f*)(orow + d) = o;
  }
}

// ---------------------------------------------------------------------------
extern "C" void kernel_launch(void* const* d_in, const int* in_sizes, int n_in,
                              void* d_out, int out_size, void* d_ws, size_t ws_size,
                              hipStream_t stream) {
  const float* x      = (const float*)d_in[0];
  const float* u      = (const float*)d_in[1];
  const float* cb     = (const float*)d_in[2];
  const float* oscale = (const float*)d_in[3];
  const float* temp   = (const float*)d_in[4];

  float* out  = (float*)d_out;
  float* soft = out + (size_t)ROWS * DIM;                       // 33554432
  float* hard = soft + (size_t)ROWS * CBN;                      // 50331648
  float* ent  = hard + (size_t)ROWS * CBN;                      // 67108864
  float* tau  = ent + 1;                                        // 67108865

  _Float16* cbn = (_Float16*)d_ws;                              // 1 MB scratch

  cb_prep_kernel<<<CBN, 256, 0, stream>>>(cb, cbn, temp, ent, tau);
  bottleneck_kernel<<<ROWS / 32, 256, 0, stream>>>(
      x, u, cb, cbn, oscale, temp, out, soft, hard, ent);
}
